// BloomGuidedCrossAttention_59691455479925
// MI455X (gfx1250) — matrix operand order repaired and checked by reading, not compiled, not run on hardware
//
#include <hip/hip_runtime.h>
#include <math.h>

// ---------------------------------------------------------------------------
// BloomGuidedCrossAttention for MI455X (gfx1250, wave32, WMMA bf16 16x16x32)
//
// Sizes: B=8, S=1024, D=768, Dbloom=64, H=4, Dh=192.  ~56 GFLOP total, attn
// output write = 134MB (~5.8us @ 23.3 TB/s) -> matrix-compute bound, so all
// GEMMs use v_wmma_f32_16x16x32_bf16 with f32 accumulation.
// GEMM: 128x128 block tile, 8 waves of 32x64 (8 WMMAs / K-step / wave),
// async global->LDS staging (ASYNCcnt) + L2 prefetch.
// ---------------------------------------------------------------------------

typedef __attribute__((ext_vector_type(16))) __bf16 v16bf;
typedef __attribute__((ext_vector_type(8)))  float  v8f;
typedef __attribute__((ext_vector_type(4)))  int    v4i;

#define BATCH   8
#define SEQ     1024
#define DMODEL  768
#define DBLOOM  64
#define NHEADS  4
#define HEADDIM 192
#define NTOK    (BATCH * SEQ)          // 8192
#define Y_ELEMS (NTOK * DMODEL)        // 6291456

// ---- gfx1250 async global->LDS staging (ASYNCcnt-tracked) -----------------
#if defined(__has_builtin)
#  if __has_builtin(__builtin_amdgcn_global_load_async_to_lds_b128) && \
      __has_builtin(__builtin_amdgcn_s_wait_asynccnt)
#    define USE_ASYNC_LDS 1
#  endif
#endif
#ifndef USE_ASYNC_LDS
#  define USE_ASYNC_LDS 0
#endif

__device__ __forceinline__ void stage16(void* lds, const void* g) {
#if USE_ASYNC_LDS
    __builtin_amdgcn_global_load_async_to_lds_b128(
        (__attribute__((address_space(1))) v4i*)g,
        (__attribute__((address_space(3))) v4i*)lds, 0, 0);
#else
    *(uint4*)lds = *(const uint4*)g;
#endif
}
__device__ __forceinline__ void stage_wait() {
#if USE_ASYNC_LDS
    __builtin_amdgcn_s_wait_asynccnt(0);
#endif
}

// ---------------- WMMA fragment helpers (per CDNA5 ISA 7.12.2) -------------
// A (16x32 bf16, row-major, stride ld): lane m = lane&15, kb = (lane>>4)*8;
// elements K = [kb..kb+7] then [kb+16..kb+23]  -> two 16B loads.
__device__ __forceinline__ v16bf load_frag_a(const __bf16* tile, int ld, int lane) {
    int m  = lane & 15;
    int kb = (lane >> 4) << 3;
    const __bf16* p = tile + (size_t)m * ld + kb;
    union { v16bf v; uint4 q[2]; } u;
    u.q[0] = *(const uint4*)(p);
    u.q[1] = *(const uint4*)(p + 16);
    return u.v;
}

// A fragment sourced from an f32 tile (softmax probs in LDS), cvt to bf16.
__device__ __forceinline__ v16bf load_frag_a_f32(const float* tile, int ld, int lane) {
    int m  = lane & 15;
    int kb = (lane >> 4) << 3;
    const float* p = tile + (size_t)m * ld + kb;
    v16bf v;
#pragma unroll
    for (int i = 0; i < 8; ++i) v[i] = (__bf16)p[i];
#pragma unroll
    for (int i = 0; i < 8; ++i) v[8 + i] = (__bf16)p[16 + i];
    return v;
}

// B supplied transposed (Bt = N x K row-major, stride ld): lane n = lane&15,
// kb = (lane>>4)*8; elements K = [kb..kb+7] then [kb+16..kb+23] -> two 16B loads.
__device__ __forceinline__ v16bf load_frag_b(const __bf16* tile, int ld, int lane) {
    int n  = lane & 15;
    int kb = (lane >> 4) << 3;
    const __bf16* p = tile + (size_t)n * ld + kb;
    union { v16bf v; uint4 q[2]; } u;
    u.q[0] = *(const uint4*)(p);
    u.q[1] = *(const uint4*)(p + 16);
    return u.v;
}

__device__ __forceinline__ v8f wmma_bf16(v16bf a, v16bf b, v8f c) {
    return __builtin_amdgcn_wmma_f32_16x16x32_bf16(
        /*neg_a=*/false, a, /*neg_b=*/false, b,
        /*c_mod=*/(short)0, c, /*reuse_a=*/false, /*reuse_b=*/false);
}

// ---------------- elementwise conversion / transpose ------------------------
__global__ void cvt_bf16_kernel(const float* __restrict__ src,
                                __bf16* __restrict__ dst, int n) {
    int i = blockIdx.x * 256 + threadIdx.x;
    if (i < n) dst[i] = (__bf16)src[i];
}

// dst[N][K] = (bf16) src[K][N]
__global__ void transpose_cvt_kernel(const float* __restrict__ src,
                                     __bf16* __restrict__ dst, int K, int N) {
    int i = blockIdx.x * 256 + threadIdx.x;
    if (i >= K * N) return;
    int n = i / K, k = i - n * K;
    dst[i] = (__bf16)src[(size_t)k * N + n];
}

// ---------------- bias MLP: gelu(bloom @ Wb1 + bb1) @ Wb2 + bb2 -------------
__global__ void bias_mlp_kernel(const float* __restrict__ bloom,
                                const float* __restrict__ Wb1, const float* __restrict__ bb1,
                                const float* __restrict__ Wb2, const float* __restrict__ bb2,
                                float* __restrict__ biasB /* [B][H][S] */) {
    int t = blockIdx.x * 256 + threadIdx.x;
    if (t >= NTOK) return;
    const float* e = bloom + (size_t)t * DBLOOM;
    float h[8];
#pragma unroll
    for (int j = 0; j < 8; ++j) h[j] = bb1[j];
    for (int k = 0; k < DBLOOM; ++k) {
        float ev = e[k];
#pragma unroll
        for (int j = 0; j < 8; ++j) h[j] += ev * Wb1[k * 8 + j];
    }
#pragma unroll
    for (int j = 0; j < 8; ++j) {
        float x = h[j];
        h[j] = 0.5f * x * (1.0f + erff(x * 0.70710678118654752f));
    }
    int b = t >> 10, s = t & 1023;
#pragma unroll
    for (int hh = 0; hh < NHEADS; ++hh) {
        float o = bb2[hh];
#pragma unroll
        for (int j = 0; j < 8; ++j) o += h[j] * Wb2[j * NHEADS + hh];
        biasB[(((b * NHEADS + hh) << 10)) + s] = o;
    }
}

// ---------------- tiled WMMA GEMM: C[M,N] = A[M,K] @ Bt[N,K]^T + bias -------
// Block = 256 threads (8 waves), tile 128(M) x 128(N), K step 32.
// Waves in a 4(M) x 2(N) grid; each wave owns 32x64 = 2x4 WMMA tiles.
// mode 0: out bf16 row-major [m*ldo + n]
// mode 1: out bf16 transposed per batch: Vt[b][n][s], b = m>>10, s = m&1023
// mode 2: out f32 row-major with residual add (for out-proj + skip connection)
__global__ __launch_bounds__(256) void gemm_bf16_kernel(
        const __bf16* __restrict__ A, const __bf16* __restrict__ Bt,
        const float* __restrict__ bias, void* __restrict__ out,
        const float* __restrict__ residual, int K, int ldo, int mode) {
    __shared__ __bf16 sA[128 * 32];
    __shared__ __bf16 sB[128 * 32];
    int tid = threadIdx.x, lane = tid & 31, wave = tid >> 5;
    int wm = wave >> 1, wn = wave & 1;                 // 4 x 2 wave grid
    int m0 = blockIdx.x * 128, n0 = blockIdx.y * 128;

    v8f acc[2][4];
#pragma unroll
    for (int i = 0; i < 2; ++i)
#pragma unroll
        for (int j = 0; j < 4; ++j)
#pragma unroll
            for (int r = 0; r < 8; ++r) acc[i][j][r] = 0.0f;

    // per-thread staging coordinates: 512 x 16B chunks per 128x32 tile
    int srow0 = tid >> 1;                  // rows 0..127
    int sch0  = (tid & 1) * 16;            // elems 0 / 16
    int srow1 = (tid + 256) >> 1;
    int sch1  = ((tid + 256) & 1) * 16;

    for (int k0 = 0; k0 < K; k0 += 32) {
        stage16(sA + srow0 * 32 + sch0, A + (size_t)(m0 + srow0) * K + k0 + sch0);
        stage16(sA + srow1 * 32 + sch1, A + (size_t)(m0 + srow1) * K + k0 + sch1);
        stage16(sB + srow0 * 32 + sch0, Bt + (size_t)(n0 + srow0) * K + k0 + sch0);
        stage16(sB + srow1 * 32 + sch1, Bt + (size_t)(n0 + srow1) * K + k0 + sch1);
        if (k0 + 32 < K) {   // L2 prefetch of next K-tile (global_prefetch_b8)
            __builtin_prefetch(A + (size_t)(m0 + srow0) * K + k0 + 32 + sch0, 0, 1);
            __builtin_prefetch(Bt + (size_t)(n0 + srow0) * K + k0 + 32 + sch0, 0, 1);
        }
        stage_wait();
        __syncthreads();

        v16bf af[2], bfv[4];
#pragma unroll
        for (int i = 0; i < 2; ++i)
            af[i] = load_frag_a(sA + (wm * 32 + i * 16) * 32, 32, lane);
#pragma unroll
        for (int j = 0; j < 4; ++j)
            bfv[j] = load_frag_b(sB + (wn * 64 + j * 16) * 32, 32, lane);
#pragma unroll
        for (int i = 0; i < 2; ++i)
#pragma unroll
            for (int j = 0; j < 4; ++j)
                acc[i][j] = wmma_bf16(af[i], bfv[j], acc[i][j]);
        __syncthreads();
    }

    int lN = lane & 15, mh = (lane >> 4) * 8;
#pragma unroll
    for (int i = 0; i < 2; ++i)
#pragma unroll
        for (int j = 0; j < 4; ++j) {
            int gn  = n0 + wn * 64 + j * 16 + lN;
            int gm0 = m0 + wm * 32 + i * 16 + mh;
            float bv = bias ? bias[gn] : 0.0f;
            if (mode == 0) {
                __bf16* o = (__bf16*)out;
#pragma unroll
                for (int r = 0; r < 8; ++r)
                    o[(size_t)(gm0 + r) * ldo + gn] = (__bf16)(acc[i][j][r] + bv);
            } else if (mode == 1) {
                __bf16 tmp[8];
#pragma unroll
                for (int r = 0; r < 8; ++r) tmp[r] = (__bf16)(acc[i][j][r] + bv);
                int b = gm0 >> 10, s = gm0 & 1023;
                __bf16* o = (__bf16*)out + (size_t)b * DMODEL * SEQ +
                            (size_t)gn * SEQ + s;
                *(uint4*)o = *(uint4*)tmp;   // 8 consecutive s values, 16B aligned
            } else {
                float* o = (float*)out;
#pragma unroll
                for (int r = 0; r < 8; ++r) {
                    size_t off = (size_t)(gm0 + r) * ldo + gn;
                    o[off] = acc[i][j][r] + bv + residual[off];
                }
            }
        }
}

// ---------------- attention: scores -> softmax(+bias) -> attn out -> ctx ----
// grid = (64 qtiles, 4 heads, 8 batches), block = 128 threads (4 waves)
#define SLD 1032   // padded f32 row stride (16B aligned, bank-skewing)
__global__ __launch_bounds__(128) void attn_kernel(
        const __bf16* __restrict__ Qb,  // [8192][768]
        const __bf16* __restrict__ Kb,  // [8192][768]
        const __bf16* __restrict__ Vt,  // [B][768][1024]
        const float* __restrict__ biasB,// [B][H][S]
        float* __restrict__ attn_out,   // [B][H][S][S]
        __bf16* __restrict__ ctxb) {    // [8192][768]
    __shared__ float sS[16 * SLD];
    __shared__ float red[128];
    int qt = blockIdx.x, h = blockIdx.y, b = blockIdx.z;
    int tid = threadIdx.x, lane = tid & 31, wave = tid >> 5;
    int lN = lane & 15, mh = (lane >> 4) * 8;
    const float scale = 0.0721687836487032f;   // 1/sqrt(192)

    // --- Q fragments for this 16-query tile (192 = 6 x 32 along K) ---
    const __bf16* Qbase = Qb + ((size_t)(b * SEQ + qt * 16) * DMODEL + h * HEADDIM);
    v16bf qf[6];
#pragma unroll
    for (int j = 0; j < 6; ++j) qf[j] = load_frag_a(Qbase + j * 32, DMODEL, lane);

    // --- scores = Q K^T * scale + bias; waves split the 64 key tiles ---
    for (int kt = wave; kt < 64; kt += 4) {
        const __bf16* Kbase = Kb + ((size_t)(b * SEQ + kt * 16) * DMODEL + h * HEADDIM);
        v8f acc;
#pragma unroll
        for (int r = 0; r < 8; ++r) acc[r] = 0.0f;
#pragma unroll
        for (int j = 0; j < 6; ++j) {
            v16bf kf = load_frag_b(Kbase + j * 32, DMODEL, lane);
            acc = wmma_bf16(qf[j], kf, acc);
        }
        float bv = biasB[((b * NHEADS + h) << 10) + kt * 16 + lN];
#pragma unroll
        for (int r = 0; r < 8; ++r)
            sS[(mh + r) * SLD + kt * 16 + lN] = acc[r] * scale + bv;
    }
    __syncthreads();

    // --- row softmax (block-wide, 16 rows), write attn f32, probs in LDS ---
    for (int r = 0; r < 16; ++r) {
        float* row = sS + r * SLD;
        float mx = -1e30f;
        for (int k = tid; k < SEQ; k += 128) mx = fmaxf(mx, row[k]);
        red[tid] = mx; __syncthreads();
        for (int s = 64; s > 0; s >>= 1) {
            if (tid < s) red[tid] = fmaxf(red[tid], red[tid + s]);
            __syncthreads();
        }
        mx = red[0]; __syncthreads();
        float sum = 0.0f;
        for (int k = tid; k < SEQ; k += 128) sum += __expf(row[k] - mx);
        red[tid] = sum; __syncthreads();
        for (int s = 64; s > 0; s >>= 1) {
            if (tid < s) red[tid] += red[tid + s];
            __syncthreads();
        }
        float inv = 1.0f / red[0]; __syncthreads();
        float* arow = attn_out +
            ((size_t)((b * NHEADS + h) * SEQ + qt * 16 + r)) * SEQ;
        for (int k = tid; k < SEQ; k += 128) {
            float p = __expf(row[k] - mx) * inv;
            row[k] = p;
            arow[k] = p;
        }
        __syncthreads();
    }

    // --- ctx = P @ V; each wave owns 3 of the 12 (16-wide) head-dim tiles ---
    v8f cacc[3];
#pragma unroll
    for (int t = 0; t < 3; ++t)
#pragma unroll
        for (int r = 0; r < 8; ++r) cacc[t][r] = 0.0f;
    const __bf16* Vb0 = Vt + (size_t)b * DMODEL * SEQ +
                        (size_t)(h * HEADDIM + wave * 48) * SEQ;
    for (int kk = 0; kk < 32; ++kk) {                 // 1024 keys / 32
        v16bf pf = load_frag_a_f32(sS + kk * 32, SLD, lane);
#pragma unroll
        for (int t = 0; t < 3; ++t) {
            v16bf vf = load_frag_b(Vb0 + (size_t)t * 16 * SEQ + kk * 32, SEQ, lane);
            cacc[t] = wmma_bf16(pf, vf, cacc[t]);
        }
    }
    __bf16* obase = ctxb + ((size_t)(b * SEQ + qt * 16) * DMODEL) + h * HEADDIM;
#pragma unroll
    for (int t = 0; t < 3; ++t) {
        int nt = wave * 3 + t;
#pragma unroll
        for (int r = 0; r < 8; ++r)
            obase[(size_t)(mh + r) * DMODEL + nt * 16 + lN] = (__bf16)cacc[t][r];
    }
}

// ---------------- per-token LayerNorm ---------------------------------------
__global__ __launch_bounds__(256) void ln_kernel(
        const float* __restrict__ x, const float* __restrict__ g,
        const float* __restrict__ bta, float* __restrict__ y) {
    __shared__ float red[256];
    int t = blockIdx.x, tid = threadIdx.x;
    const float* row = x + (size_t)t * DMODEL;
    float s = 0.0f;
    for (int k = tid; k < DMODEL; k += 256) s += row[k];
    red[tid] = s; __syncthreads();
    for (int q = 128; q > 0; q >>= 1) {
        if (tid < q) red[tid] += red[tid + q];
        __syncthreads();
    }
    float mu = red[0] * (1.0f / DMODEL); __syncthreads();
    float v = 0.0f;
    for (int k = tid; k < DMODEL; k += 256) { float d = row[k] - mu; v += d * d; }
    red[tid] = v; __syncthreads();
    for (int q = 128; q > 0; q >>= 1) {
        if (tid < q) red[tid] += red[tid + q];
        __syncthreads();
    }
    float inv = rsqrtf(red[0] * (1.0f / DMODEL) + 1e-5f);
    float* yr = y + (size_t)t * DMODEL;
    for (int k = tid; k < DMODEL; k += 256)
        yr[k] = (row[k] - mu) * inv * g[k] + bta[k];
}

// ---------------------------------------------------------------------------
extern "C" void kernel_launch(void* const* d_in, const int* in_sizes, int n_in,
                              void* d_out, int out_size, void* d_ws, size_t ws_size,
                              hipStream_t stream) {
    const float* X     = (const float*)d_in[0];   // dnabert_hidden [8,1024,768]
    const float* E     = (const float*)d_in[1];   // bloom_encoding [8,1024,64]
    const float* Wq    = (const float*)d_in[2];
    const float* bq    = (const float*)d_in[3];
    const float* Wk    = (const float*)d_in[4];
    const float* bk    = (const float*)d_in[5];
    const float* Wv    = (const float*)d_in[6];
    const float* bv    = (const float*)d_in[7];
    const float* Wb1   = (const float*)d_in[8];
    const float* bb1   = (const float*)d_in[9];
    const float* Wb2   = (const float*)d_in[10];
    const float* bb2   = (const float*)d_in[11];
    const float* Wo    = (const float*)d_in[12];
    const float* bo    = (const float*)d_in[13];
    const float* ln_g  = (const float*)d_in[14];
    const float* ln_b  = (const float*)d_in[15];

    // workspace carve-up (256B aligned)
    char* ws = (char*)d_ws;
    size_t off = 0;
    auto carve = [&](size_t bytes) { char* p = ws + off; off += (bytes + 255) & ~(size_t)255; return p; };
    __bf16* Xb    = (__bf16*)carve((size_t)NTOK * DMODEL * 2);
    __bf16* Eb    = (__bf16*)carve((size_t)NTOK * DBLOOM * 2);
    __bf16* WqT   = (__bf16*)carve((size_t)DMODEL * DMODEL * 2);
    __bf16* WkT   = (__bf16*)carve((size_t)DMODEL * DBLOOM * 2);
    __bf16* WvT   = (__bf16*)carve((size_t)DMODEL * DMODEL * 2);
    __bf16* WoT   = (__bf16*)carve((size_t)DMODEL * DMODEL * 2);
    __bf16* Qb    = (__bf16*)carve((size_t)NTOK * DMODEL * 2);
    __bf16* Kb    = (__bf16*)carve((size_t)NTOK * DMODEL * 2);
    __bf16* Vt    = (__bf16*)carve((size_t)BATCH * DMODEL * SEQ * 2);
    float*  biasB = (float*) carve((size_t)BATCH * NHEADS * SEQ * 4);
    __bf16* ctxb  = (__bf16*)carve((size_t)NTOK * DMODEL * 2);
    float*  xbuf  = (float*) carve((size_t)NTOK * DMODEL * 4);

    float* y_out    = (float*)d_out;
    float* attn_out = (float*)d_out + Y_ELEMS;

    // 1) activation conversion + weight transpose-conversion
    cvt_bf16_kernel<<<(NTOK * DMODEL + 255) / 256, 256, 0, stream>>>(X, Xb, NTOK * DMODEL);
    cvt_bf16_kernel<<<(NTOK * DBLOOM + 255) / 256, 256, 0, stream>>>(E, Eb, NTOK * DBLOOM);
    transpose_cvt_kernel<<<(DMODEL * DMODEL + 255) / 256, 256, 0, stream>>>(Wq, WqT, DMODEL, DMODEL);
    transpose_cvt_kernel<<<(DBLOOM * DMODEL + 255) / 256, 256, 0, stream>>>(Wk, WkT, DBLOOM, DMODEL);
    transpose_cvt_kernel<<<(DMODEL * DMODEL + 255) / 256, 256, 0, stream>>>(Wv, WvT, DMODEL, DMODEL);
    transpose_cvt_kernel<<<(DMODEL * DMODEL + 255) / 256, 256, 0, stream>>>(Wo, WoT, DMODEL, DMODEL);

    // 2) bias MLP
    bias_mlp_kernel<<<NTOK / 256, 256, 0, stream>>>(E, Wb1, bb1, Wb2, bb2, biasB);

    // 3) Q/K/V projections (WMMA GEMMs); V stored transposed per batch
    dim3 ggrid(NTOK / 128, DMODEL / 128);
    gemm_bf16_kernel<<<ggrid, 256, 0, stream>>>(Xb, WqT, bq, (void*)Qb, nullptr, DMODEL, DMODEL, 0);
    gemm_bf16_kernel<<<ggrid, 256, 0, stream>>>(Eb, WkT, bk, (void*)Kb, nullptr, DBLOOM, DMODEL, 0);
    gemm_bf16_kernel<<<ggrid, 256, 0, stream>>>(Xb, WvT, bv, (void*)Vt, nullptr, DMODEL, 0, 1);

    // 4) attention (scores -> softmax+bias -> attn out -> ctx)
    attn_kernel<<<dim3(SEQ / 16, NHEADS, BATCH), 128, 0, stream>>>(
        Qb, Kb, Vt, biasB, attn_out, ctxb);

    // 5) out projection + residual, then LayerNorm
    gemm_bf16_kernel<<<ggrid, 256, 0, stream>>>(ctxb, WoT, bo, (void*)xbuf, X, DMODEL, DMODEL, 2);
    ln_kernel<<<NTOK, 256, 0, stream>>>(xbuf, ln_g, ln_b, y_out);
}
